// DFEM_59407987638548
// MI455X (gfx1250) — compile-verified, matrix operand order checked by reading, not run on hardware
//
#include <hip/hip_runtime.h>
#include <stdint.h>

#define DEVINL __device__ __forceinline__

typedef __attribute__((ext_vector_type(16))) __bf16 v16bf;
typedef __attribute__((ext_vector_type(8)))  float   v8f;
typedef __attribute__((ext_vector_type(8)))  unsigned int v8u;
typedef __attribute__((ext_vector_type(4)))  unsigned int u32x4;
typedef __attribute__((ext_vector_type(8)))  int i32x8;
typedef __attribute__((ext_vector_type(4)))  int i32x4;

#if __has_builtin(__builtin_amdgcn_tensor_load_to_lds) && \
    __has_builtin(__builtin_amdgcn_s_wait_tensorcnt)
#define USE_TDM 1
#else
#define USE_TDM 0
#endif

#if __has_builtin(__builtin_amdgcn_update_dpp)
#define USE_DPP 1
#else
#define USE_DPP 0
#endif

constexpr int Bsz = 8;
constexpr int C   = 256;
constexpr int Hh  = 64;
constexpr int Ww  = 64;
constexpr int N   = Hh * Ww;   // 4096
constexpr int C8  = 32;        // q/k channels

// ---------- helpers ----------
DEVINL unsigned short f2bf(float f) {
  unsigned int u = __float_as_uint(f);
  unsigned int r = u + 0x7FFFu + ((u >> 16) & 1u);   // RNE (NaN edge ignored)
  return (unsigned short)(r >> 16);
}

DEVINL v16bf pack_ab(uint4 a, uint4 b) {
  v8u u;
  u[0] = a.x; u[1] = a.y; u[2] = a.z; u[3] = a.w;
  u[4] = b.x; u[5] = b.y; u[6] = b.z; u[7] = b.w;
  return __builtin_bit_cast(v16bf, u);
}

// ---- 16-lane butterfly reductions via DPP16 (no LDS traffic) ----
#if USE_DPP
template <int CTRL>
DEVINL float dpp_mov(float v) {
  return __int_as_float(__builtin_amdgcn_update_dpp(
      0, __float_as_int(v), CTRL, 0xF, 0xF, true));
}
DEVINL float red_max16(float v) {
  v = fmaxf(v, dpp_mov<0xB1>(v));    // quad_perm [1,0,3,2]  (xor 1)
  v = fmaxf(v, dpp_mov<0x4E>(v));    // quad_perm [2,3,0,1]  (xor 2)
  v = fmaxf(v, dpp_mov<0x141>(v));   // row_half_mirror      (other quad in 8)
  v = fmaxf(v, dpp_mov<0x140>(v));   // row_mirror           (other 8 in 16)
  return v;
}
DEVINL float red_sum16(float v) {
  v = v + dpp_mov<0xB1>(v);
  v = v + dpp_mov<0x4E>(v);
  v = v + dpp_mov<0x141>(v);
  v = v + dpp_mov<0x140>(v);
  return v;
}
#else
DEVINL float red_max16(float v) {
  for (int msk = 1; msk < 16; msk <<= 1) v = fmaxf(v, __shfl_xor(v, msk, 32));
  return v;
}
DEVINL float red_sum16(float v) {
  for (int msk = 1; msk < 16; msk <<= 1) v += __shfl_xor(v, msk, 32);
  return v;
}
#endif

#if USE_TDM
// Issue a 2-D TDM load (data_size = 2B) with LDS padding of 16 B after every
// 64 B stored -> LDS row pitch 80 B (40 shorts), matching our operand layout.
DEVINL void tdm_load_2d(unsigned lds_off, const void* gptr,
                        unsigned dim0, unsigned dim1,
                        unsigned tile0, unsigned tile1, unsigned stride) {
  unsigned long long ga = (unsigned long long)(size_t)gptr;
  u32x4 g0;
  g0[0] = 1u;                                    // count=1, user descriptor
  g0[1] = lds_off;                               // LDS dest (bytes)
  g0[2] = (unsigned)ga;                          // global addr lo
  g0[3] = (unsigned)((ga >> 32) & 0x1ffffffu)    // global addr hi
          | (2u << 30);                          // type = 2 (image)
  i32x8 g1;
  g1[0] = (int)((1u << 16)      // data_size = 2 bytes
              | (1u << 20)      // pad_enable
              | (3u << 22)      // pad_interval: 16 DWORDs (64 B)
              | (3u << 25));    // pad_amount : 4 DWORDs (16 B)
  g1[1] = (int)((dim0 & 0xffffu) << 16);                       // dim0 lo16
  g1[2] = (int)(((dim0 >> 16) & 0xffffu) | ((dim1 & 0xffffu) << 16));
  g1[3] = (int)(((dim1 >> 16) & 0xffffu) | (tile0 << 16));     // tile_dim0
  g1[4] = (int)(tile1 & 0xffffu);                              // tile_dim1
  g1[5] = (int)stride;                                         // dim0 stride
  g1[6] = 0;
  g1[7] = 0;
  i32x4 zz4;
  zz4[0] = 0; zz4[1] = 0; zz4[2] = 0; zz4[3] = 0;
  i32x8 zz8;
#pragma unroll
  for (int q = 0; q < 8; ++q) zz8[q] = 0;
  __builtin_amdgcn_tensor_load_to_lds(g0, g1, zz4, zz4, zz8, 0);
}
#endif

// ---------- f32 -> bf16 convert ----------
__global__ void cvt_f32_bf16(const float* __restrict__ in,
                             unsigned short* __restrict__ out, int n) {
  int i = blockIdx.x * blockDim.x + threadIdx.x;
  if (i < n) out[i] = f2bf(in[i]);
}

// ---------- generic bf16 WMMA GEMM ----------
// C[z] = A[M,K] * B[z][K,Nc] + bias. Compile-time output selection:
//   HASF : f32  [z][M][Nc]
//   HASB : bf16 [z][M][Nc]
//   HAST : bf16 [z][Nc][M]  (transposed)
template <bool HASF, bool HASB, bool HAST>
__global__ __launch_bounds__(128) void gemm_bf16(
    const unsigned short* __restrict__ A,
    const unsigned short* __restrict__ Bg,
    const float* __restrict__ bias,
    float* __restrict__ Cf,
    unsigned short* __restrict__ Cb,
    unsigned short* __restrict__ Ct,
    int M, int K, int Nc)
{
  // B chunk staged TRANSPOSED: [n][k], pitch 40 shorts (80 B, 16B-aligned rows)
  __shared__ unsigned short bl[64 * 40];

  const int z    = blockIdx.z;
  const int tid  = threadIdx.x;
  const int wv   = tid >> 5;
  const int lane = tid & 31;
  const int l16  = lane & 15;
  const int hi   = lane >> 4;
  const int lo   = hi ? 8 : 0;
  const int n0   = blockIdx.x * 64;
  const int m0   = blockIdx.y * 64 + wv * 16;
  const bool act = (m0 < M);

  const unsigned short* Bz = Bg + (size_t)z * K * Nc;

  v8f acc[4];
#pragma unroll
  for (int t = 0; t < 4; ++t)
#pragma unroll
    for (int r = 0; r < 8; ++r) acc[t][r] = 0.f;

  for (int k0 = 0; k0 < K; k0 += 32) {
    __syncthreads();
    {  // cooperative stage of B chunk [32 k][64 n], transposed into LDS [n][k]
      int kk = tid >> 2, c0 = (tid & 3) * 16;
      const unsigned short* srow = Bz + (size_t)(k0 + kk) * Nc + n0 + c0;
      const uint4* src = (const uint4*)srow;
      uint4 q0 = src[0];
      uint4 q1 = src[1];
      if (k0 + 32 < K)
        __builtin_prefetch(srow + (size_t)32 * Nc, 0, 1);
      unsigned w[8] = {q0.x, q0.y, q0.z, q0.w, q1.x, q1.y, q1.z, q1.w};
#pragma unroll
      for (int j = 0; j < 8; ++j) {
        bl[(c0 + 2 * j + 0) * 40 + kk] = (unsigned short)(w[j] & 0xffffu);
        bl[(c0 + 2 * j + 1) * 40 + kk] = (unsigned short)(w[j] >> 16);
      }
    }
    __syncthreads();
    if (act) {
      const unsigned short* arow = A + (size_t)(m0 + l16) * K + k0;
      v16bf av = pack_ab(*(const uint4*)(arow + lo),
                         *(const uint4*)(arow + 16 + lo));
#pragma unroll
      for (int t = 0; t < 4; ++t) {
        const unsigned short* br = &bl[(t * 16 + l16) * 40];
        v16bf bv = pack_ab(*(const uint4*)(br + lo),
                           *(const uint4*)(br + 16 + lo));
        acc[t] = __builtin_amdgcn_wmma_f32_16x16x32_bf16(
            false, av, false, bv, (short)0, acc[t], false, false);
      }
    }
  }

  if (act) {
    float bv8[8];
#pragma unroll
    for (int r = 0; r < 8; ++r) bv8[r] = bias[m0 + 8 * hi + r];
#pragma unroll
    for (int t = 0; t < 4; ++t) {
      int n = n0 + t * 16 + l16;
      float vals[8];
#pragma unroll
      for (int r = 0; r < 8; ++r) vals[r] = acc[t][r] + bv8[r];
      if constexpr (HASF) {
#pragma unroll
        for (int r = 0; r < 8; ++r)
          Cf[(size_t)z * M * Nc + (size_t)(m0 + 8 * hi + r) * Nc + n] = vals[r];
      }
      if constexpr (HASB) {
#pragma unroll
        for (int r = 0; r < 8; ++r)
          Cb[(size_t)z * M * Nc + (size_t)(m0 + 8 * hi + r) * Nc + n] =
              f2bf(vals[r]);
      }
      if constexpr (HAST) {
        uint4 o;
        o.x = (unsigned)f2bf(vals[0]) | ((unsigned)f2bf(vals[1]) << 16);
        o.y = (unsigned)f2bf(vals[2]) | ((unsigned)f2bf(vals[3]) << 16);
        o.z = (unsigned)f2bf(vals[4]) | ((unsigned)f2bf(vals[5]) << 16);
        o.w = (unsigned)f2bf(vals[6]) | ((unsigned)f2bf(vals[7]) << 16);
        *(uint4*)(Ct + (size_t)z * Nc * M + (size_t)n * M + m0 + 8 * hi) = o;
      }
    }
  }
}

// ---------- spatial attention: channel mean/max over both streams ----------
__global__ void sa_maps(const float* __restrict__ X, float* __restrict__ maps) {
  int idx = blockIdx.x * blockDim.x + threadIdx.x;  // b*N + n
  if (idx >= Bsz * N) return;
  int b = idx / N, n = idx - b * N;
  float sum = 0.f, mx = -3.0e38f;
  for (int s = 0; s < 2; ++s) {
    const float* base = X + (size_t)(s * Bsz + b) * C * N + n;
    for (int c = 0; c < C; ++c) {
      float v = base[(size_t)c * N];
      sum += v;
      mx = fmaxf(mx, v);
    }
  }
  maps[(size_t)(b * 2 + 0) * N + n] = sum * (1.0f / (2 * C));
  maps[(size_t)(b * 2 + 1) * N + n] = mx;
}

// ---------- 3x3 conv (2->1 ch, zero pad) + sigmoid ----------
__global__ void sa_conv(const float* __restrict__ maps,
                        const float* __restrict__ wsa,
                        float* __restrict__ wt) {
  int idx = blockIdx.x * blockDim.x + threadIdx.x;
  if (idx >= Bsz * N) return;
  int b = idx / N, n = idx - b * N;
  int y = n >> 6, x = n & 63;
  float acc = 0.f;
  for (int i = 0; i < 2; ++i)
    for (int dy = 0; dy < 3; ++dy) {
      int yy = y + dy - 1;
      if (yy < 0 || yy >= Hh) continue;
      for (int dx = 0; dx < 3; ++dx) {
        int xx = x + dx - 1;
        if (xx < 0 || xx >= Ww) continue;
        acc += wsa[i * 9 + dy * 3 + dx] *
               maps[(size_t)(b * 2 + i) * N + yy * Ww + xx];
      }
    }
  wt[idx] = 1.0f / (1.0f + __expf(-acc));
}

// ---------- flash-attention PAM: X <- gamma * softmax(q^T k) v + X ----------
// QT,KT: bf16 [z][N][32]; Vb: bf16 [z][C][N]; X: f32 [z][C][N] (in-place)
__global__ __launch_bounds__(128) void flash_pam(
    const unsigned short* __restrict__ QT,
    const unsigned short* __restrict__ KT,
    const unsigned short* __restrict__ Vb,
    float* __restrict__ X,
    const float* __restrict__ gamma)
{
  __shared__ unsigned short kl[2][32 * 40];     // [key][feat], 80 B pitch
  __shared__ unsigned short vl[2][256 * 40];    // [ch][key],   80 B pitch
  __shared__ unsigned short pl[4][16 * 32];     // per-wave P transpose buf

  const int z    = blockIdx.z;
  const int tid  = threadIdx.x;
  const int wv   = tid >> 5;
  const int lane = tid & 31;
  const int l16  = lane & 15;
  const int hi   = lane >> 4;
  const int lo   = hi ? 8 : 0;
  const int i0   = blockIdx.x * 64 + wv * 16;

  const unsigned short* qt = QT + (size_t)z * N * C8;
  const unsigned short* kt = KT + (size_t)z * N * C8;
  const unsigned short* vb = Vb + (size_t)z * C * N;
  float* Xz = X + (size_t)z * C * N;

  // Q tile as A operand (16 queries x 32 feats), loaded once
  const unsigned short* qrow = qt + (size_t)(i0 + l16) * C8;
  v16bf Aq = pack_ab(*(const uint4*)(qrow + lo),
                     *(const uint4*)(qrow + 16 + lo));

  v8f O[16];
  float mr[8], lr[8];
#pragma unroll
  for (int t = 0; t < 16; ++t)
#pragma unroll
    for (int r = 0; r < 8; ++r) O[t][r] = 0.f;
#pragma unroll
  for (int r = 0; r < 8; ++r) { mr[r] = -3.0e38f; lr[r] = 0.f; }

  constexpr int NIT    = N / 32;
  constexpr unsigned KB = 32 * 40 * 2;    // bytes per kl buffer
  constexpr unsigned VB = 256 * 40 * 2;   // bytes per vl buffer

#if USE_TDM
  const unsigned kl_off = (unsigned)(size_t)(&kl[0][0]);
  const unsigned vl_off = (unsigned)(size_t)(&vl[0][0]);
  if (wv == 0) {  // prologue: stage chunk 0 into buffer 0 via TDM
    tdm_load_2d(vl_off, vb, N, C, 32, 256, N);
    tdm_load_2d(kl_off, kt, (unsigned)(N * C8), 1, 1024, 0,
                (unsigned)(N * C8));
  }
#endif

  for (int it = 0; it < NIT; ++it) {
    const int buf = it & 1;
#if USE_TDM
    if (wv == 0) __builtin_amdgcn_s_wait_tensorcnt(0);
    __syncthreads();
    if (wv == 0 && it + 1 < NIT) {  // overlap next chunk's DMA with compute
      int jn = (it + 1) * 32;
      tdm_load_2d(vl_off + (unsigned)(buf ^ 1) * VB, vb + jn,
                  N, C, 32, 256, N);
      tdm_load_2d(kl_off + (unsigned)(buf ^ 1) * KB,
                  kt + (size_t)jn * C8,
                  (unsigned)(N * C8), 1, 1024, 0, (unsigned)(N * C8));
    }
#else
    __syncthreads();
    {  // fallback: cooperative vector-load staging
      int j0 = it * 32;
      int key = tid >> 2, c4 = (tid & 3);
      *(uint4*)(&kl[buf][key * 40 + c4 * 8]) =
          *(const uint4*)(kt + (size_t)(j0 + key) * C8 + c4 * 8);
      int c0 = tid * 2;
      const uint4* s0 = (const uint4*)(vb + (size_t)c0 * N + j0);
      const uint4* s1 = (const uint4*)(vb + (size_t)(c0 + 1) * N + j0);
      uint4* d0 = (uint4*)(&vl[buf][c0 * 40]);
      uint4* d1 = (uint4*)(&vl[buf][(c0 + 1) * 40]);
      d0[0] = s0[0]; d0[1] = s0[1]; d0[2] = s0[2]; d0[3] = s0[3];
      d1[0] = s1[0]; d1[1] = s1[1]; d1[2] = s1[2]; d1[3] = s1[3];
    }
    __syncthreads();
#endif

    // S = Q^T K for 32 keys (two 16-key WMMAs)
    v8f zero;
#pragma unroll
    for (int r = 0; r < 8; ++r) zero[r] = 0.f;
    const unsigned short* kr0 = &kl[buf][l16 * 40];
    v16bf B0 = pack_ab(*(const uint4*)(kr0 + lo),
                       *(const uint4*)(kr0 + 16 + lo));
    v8f S0 = __builtin_amdgcn_wmma_f32_16x16x32_bf16(
        false, Aq, false, B0, (short)0, zero, false, false);
    const unsigned short* kr1 = &kl[buf][(16 + l16) * 40];
    v16bf B1 = pack_ab(*(const uint4*)(kr1 + lo),
                       *(const uint4*)(kr1 + 16 + lo));
    v8f S1 = __builtin_amdgcn_wmma_f32_16x16x32_bf16(
        false, Aq, false, B1, (short)0, zero, false, false);

    // online softmax (rows live across 16-lane halves in C layout);
    // 16-lane butterflies done with DPP16 (no LDS traffic)
    float sc[8];
#pragma unroll
    for (int r = 0; r < 8; ++r) {
      float cm = red_max16(fmaxf(S0[r], S1[r]));
      float mn = fmaxf(mr[r], cm);
      sc[r] = __expf(mr[r] - mn);
      mr[r] = mn;
      S0[r] = __expf(S0[r] - mn);
      S1[r] = __expf(S1[r] - mn);
      float rs = red_sum16(S0[r] + S1[r]);
      lr[r] = lr[r] * sc[r] + rs;
    }
#pragma unroll
    for (int t = 0; t < 16; ++t)
#pragma unroll
      for (int r = 0; r < 8; ++r) O[t][r] *= sc[r];

    // transpose P (C layout -> A layout) through per-wave LDS buffer
    unsigned short* pw = &pl[wv][0];
#pragma unroll
    for (int r = 0; r < 8; ++r) {
      int m = r + 8 * hi;
      pw[m * 32 + l16]      = f2bf(S0[r]);
      pw[m * 32 + 16 + l16] = f2bf(S1[r]);
    }
    // same-wave DS ops are in order; compiler inserts s_wait_dscnt
    const unsigned short* pr = &pw[l16 * 32];
    v16bf Pa = pack_ab(*(const uint4*)(pr + lo),
                       *(const uint4*)(pr + 16 + lo));

    // O += P * V  (16 channel tiles)
#pragma unroll
    for (int t = 0; t < 16; ++t) {
      const unsigned short* vr = &vl[buf][(t * 16 + l16) * 40];
      v16bf Bv = pack_ab(*(const uint4*)(vr + lo),
                         *(const uint4*)(vr + 16 + lo));
      O[t] = __builtin_amdgcn_wmma_f32_16x16x32_bf16(
          false, Pa, false, Bv, (short)0, O[t], false, false);
    }
  }

  // finalize: X <- gamma * (O / l) + X
  float g = gamma[0];
  float inv[8];
#pragma unroll
  for (int r = 0; r < 8; ++r) inv[r] = 1.0f / lr[r];
#pragma unroll
  for (int t = 0; t < 16; ++t) {
    int c = t * 16 + l16;
#pragma unroll
    for (int r = 0; r < 8; ++r) {
      size_t ad = (size_t)c * N + (i0 + r + 8 * hi);
      float v = O[t][r] * inv[r];
      Xz[ad] = g * v + Xz[ad];
    }
  }
}

// ---------- final elementwise merge ----------
__global__ void finalize_k(const float* __restrict__ x1,
                           const float* __restrict__ x2,
                           const float* __restrict__ X,
                           const float* __restrict__ wt,
                           float* __restrict__ out) {
  int idx = blockIdx.x * blockDim.x + threadIdx.x;
  if (idx >= Bsz * C * N) return;
  int b = idx / (C * N);
  int n = idx % N;
  float p1 = X[idx];
  float p2 = X[idx + (size_t)Bsz * C * N];
  float a  = x1[idx] * p1;
  float bb = x2[idx] * p2;
  out[idx] = wt[(size_t)b * N + n] * fabsf(bb - a);
}

// ---------- launch ----------
extern "C" void kernel_launch(void* const* d_in, const int* in_sizes, int n_in,
                              void* d_out, int out_size, void* d_ws,
                              size_t ws_size, hipStream_t stream) {
  const float* x1    = (const float*)d_in[0];
  const float* x2    = (const float*)d_in[1];
  const float* w1    = (const float*)d_in[2];
  const float* b1    = (const float*)d_in[3];
  const float* wq    = (const float*)d_in[4];
  const float* bq    = (const float*)d_in[5];
  const float* wk    = (const float*)d_in[6];
  const float* bk    = (const float*)d_in[7];
  const float* wvp   = (const float*)d_in[8];
  const float* bvp   = (const float*)d_in[9];
  const float* gamma = (const float*)d_in[10];
  const float* wsa   = (const float*)d_in[11];
  float* out = (float*)d_out;

  char* ws = (char*)d_ws;
  size_t off = 0;
  auto alloc = [&](size_t bytes) -> char* {
    char* p = ws + off;
    off += (bytes + 255) & ~(size_t)255;
    return p;
  };

  const size_t SCN = (size_t)2 * Bsz * C * N;  // elems in [s][b][c][n]

  float*          X      = (float*)alloc(sizeof(float) * SCN);  // x11/x21 -> PAM out
  unsigned short* Xb_in  = (unsigned short*)alloc(2 * SCN);     // bf16(x1,x2)
  unsigned short* Xb_mid = (unsigned short*)alloc(2 * SCN);     // bf16(x11,x21)
  unsigned short* QT     = (unsigned short*)alloc((size_t)2 * 2 * Bsz * N * C8);
  unsigned short* KT     = (unsigned short*)alloc((size_t)2 * 2 * Bsz * N * C8);
  unsigned short* Vb     = (unsigned short*)alloc(2 * SCN);
  float*          maps   = (float*)alloc(sizeof(float) * Bsz * 2 * N);
  float*          wt     = (float*)alloc(sizeof(float) * Bsz * N);
  unsigned short* w1b    = (unsigned short*)alloc(2 * 256 * 256);
  unsigned short* wqb    = (unsigned short*)alloc(2 * 32 * 256);
  unsigned short* wkb    = (unsigned short*)alloc(2 * 32 * 256);
  unsigned short* wvb    = (unsigned short*)alloc(2 * 256 * 256);
  (void)ws_size; (void)in_sizes; (void)n_in; (void)out_size;

  const int npix = Bsz * C * N;  // 8388608 per stream

  // 1) bf16 conversions
  cvt_f32_bf16<<<(npix + 255) / 256, 256, 0, stream>>>(x1, Xb_in, npix);
  cvt_f32_bf16<<<(npix + 255) / 256, 256, 0, stream>>>(x2, Xb_in + npix, npix);
  cvt_f32_bf16<<<(65536 + 255) / 256, 256, 0, stream>>>(w1, w1b, 256 * 256);
  cvt_f32_bf16<<<(8192 + 255) / 256, 256, 0, stream>>>(wq, wqb, 32 * 256);
  cvt_f32_bf16<<<(8192 + 255) / 256, 256, 0, stream>>>(wk, wkb, 32 * 256);
  cvt_f32_bf16<<<(65536 + 255) / 256, 256, 0, stream>>>(wvp, wvb, 256 * 256);

  // 2) x11/x21 = w1 * x + b1   (f32 + bf16 outputs)
  gemm_bf16<true, true, false><<<dim3(64, 4, 16), 128, 0, stream>>>(
      w1b, Xb_in, b1, X, Xb_mid, nullptr, 256, 256, 4096);

  // 3) q,k (transposed bf16) and v projections
  gemm_bf16<false, false, true><<<dim3(64, 1, 16), 128, 0, stream>>>(
      wqb, Xb_mid, bq, nullptr, nullptr, QT, 32, 256, 4096);
  gemm_bf16<false, false, true><<<dim3(64, 1, 16), 128, 0, stream>>>(
      wkb, Xb_mid, bk, nullptr, nullptr, KT, 32, 256, 4096);
  gemm_bf16<false, true, false><<<dim3(64, 4, 16), 128, 0, stream>>>(
      wvb, Xb_mid, bvp, nullptr, Vb, nullptr, 256, 256, 4096);

  // 4) spatial attention weight (must read X before flash overwrites it)
  sa_maps<<<(Bsz * N + 255) / 256, 256, 0, stream>>>(X, maps);
  sa_conv<<<(Bsz * N + 255) / 256, 256, 0, stream>>>(maps, wsa, wt);

  // 5) flash-attention PAM, in-place residual into X (double-buffered TDM)
  flash_pam<<<dim3(64, 1, 16), 128, 0, stream>>>(QT, KT, Vb, X, gamma);

  // 6) final merge
  finalize_k<<<(npix + 255) / 256, 256, 0, stream>>>(x1, x2, X, wt, out);
}
